// Refine_73821897884021
// MI455X (gfx1250) — matrix operand order repaired and checked by reading, not compiled, not run on hardware
//
#include <hip/hip_runtime.h>

#define CC   1024
#define HW   6144            // 64*96
#define NB   8
#define NCLU 80
#define NPIX 49152           // 8*6144

typedef __attribute__((ext_vector_type(16))) __bf16       v16bf;
typedef __attribute__((ext_vector_type(8)))  float        v8f;
typedef __attribute__((ext_vector_type(8)))  unsigned int v8u;

__device__ __forceinline__ unsigned short f2bf(float f) {
  unsigned u = __builtin_bit_cast(unsigned, f);
  u += 0x7FFFu + ((u >> 16) & 1u);          // round to nearest even
  return (unsigned short)(u >> 16);
}

__device__ __forceinline__ v8f vzero8() {
  v8f z = {0.f,0.f,0.f,0.f,0.f,0.f,0.f,0.f};
  return z;
}

// LDS byte offset of a __shared__ element: flat shared-aperture address has the
// AS(3) offset in its low 32 bits (per the ISA aperture mapping).
__device__ __forceinline__ unsigned ldsoff(const void* p) {
  return (unsigned)(unsigned long long)p;
}

// async global -> LDS copy of 16 bytes (GLOBAL_LOAD_ASYNC_TO_LDS_B128, GVS mode)
__device__ __forceinline__ void async_cp16(unsigned lds, const void* gbase, unsigned goff) {
  asm volatile("global_load_async_to_lds_b128 %0, %1, %2"
               :: "v"(lds), "v"(goff), "s"(gbase) : "memory");
}

__device__ __forceinline__ void wait_async0() {
  asm volatile("s_wait_asynccnt 0x0" ::: "memory");
}

// ---------------------------------------------------------------------------
// 1) normalize centroids -> bf16, packed as [c/2][n] uint pairs (WMMA B layout)
// ---------------------------------------------------------------------------
__global__ __launch_bounds__(256) void k_cent(const float* __restrict__ cent,
                                              unsigned int* __restrict__ cnP) {
  __shared__ float red[256];
  int n = blockIdx.x, t = threadIdx.x;
  float ss = 0.f;
  for (int c = t; c < CC; c += 256) { float v = cent[n * CC + c]; ss += v * v; }
  red[t] = ss; __syncthreads();
  for (int s = 128; s > 0; s >>= 1) { if (t < s) red[t] += red[t + s]; __syncthreads(); }
  float scale = 1.0f / fmaxf(sqrtf(red[0]), 1e-12f);
  for (int pr = t; pr < CC / 2; pr += 256) {
    int c = pr * 2;
    unsigned lo = f2bf(cent[n * CC + c] * scale);
    unsigned hi = f2bf(cent[n * CC + c + 1] * scale);
    cnP[(size_t)pr * NCLU + n] = lo | (hi << 16);
  }
}

// ---------------------------------------------------------------------------
// 2) sim = x . cn  (WMMA bf16, 128 pixels x 80 clusters per block) + argmax
// ---------------------------------------------------------------------------
__global__ __launch_bounds__(256) void k_sim(const float* __restrict__ x,
                                             const unsigned int* __restrict__ cnP,
                                             int* __restrict__ idx) {
  __shared__ unsigned int As[128][17];     // [pixel][K-pair]
  __shared__ float simS[128][NCLU];
  int t = threadIdx.x, lane = t & 31, wid = t >> 5;
  int p0 = blockIdx.x * 128;
  int b = p0 / HW, s0 = p0 % HW;
  const float* xb = x + (size_t)b * CC * HW + s0;

  v8f acc[5];
#pragma unroll
  for (int j = 0; j < 5; ++j) acc[j] = vzero8();

  for (int k0 = 0; k0 < CC; k0 += 32) {
    __syncthreads();
    {   // stage 128 pixels x 32 channels, fp32 -> bf16 pairs
      int c2 = t >> 4;                       // 0..15 -> channel pair
      int c  = k0 + 2 * c2;
      const float* xr0 = xb + (size_t)c * HW;
      const float* xr1 = xb + (size_t)(c + 1) * HW;
#pragma unroll
      for (int i = 0; i < 8; ++i) {
        int pl = (t & 15) + 16 * i;          // coalesced over pixels
        unsigned lo = f2bf(xr0[pl]);
        unsigned hi = f2bf(xr1[pl]);
        As[pl][c2] = lo | (hi << 16);
      }
    }
    __syncthreads();

    // A fragment (16x32 bf16): lanes 0-15 K pairs {0..3,8..11}, lanes 16-31 {4..7,12..15}
    v8u au;
    int m   = (wid << 4) + (lane & 15);
    int khp = (lane >> 4) * 4;
#pragma unroll
    for (int r = 0; r < 4; ++r) au[r]     = As[m][khp + r];
#pragma unroll
    for (int r = 0; r < 4; ++r) au[4 + r] = As[m][8 + khp + r];
    v16bf av = __builtin_bit_cast(v16bf, au);

    int kbp = (lane >> 4) * 4;               // B: same K pairs as A: lanes 0-15 {0..3,8..11}, 16-31 {4..7,12..15}
#pragma unroll
    for (int j = 0; j < 5; ++j) {
      v8u bu;
      int ncol = j * 16 + (lane & 15);
#pragma unroll
      for (int r = 0; r < 8; ++r)
        bu[r] = cnP[(size_t)((k0 >> 1) + kbp + (r & 3) + ((r >> 2) << 3)) * NCLU + ncol];
      v16bf bv = __builtin_bit_cast(v16bf, bu);
      acc[j] = __builtin_amdgcn_wmma_f32_16x16x32_bf16(
          false, av, false, bv, (short)0, acc[j], false, false);
    }
  }

  __syncthreads();
  {   // C layout: VGPR v -> row v (lanes<16) / v+8 (lanes>=16), col = lane&15
    int mbase = (wid << 4) + ((lane >> 4) << 3);
    int ncl = lane & 15;
#pragma unroll
    for (int j = 0; j < 5; ++j)
#pragma unroll
      for (int v = 0; v < 8; ++v)
        simS[mbase + v][j * 16 + ncl] = acc[j][v];
  }
  __syncthreads();
  if (t < 128) {
    float best = simS[t][0]; int bi = 0;
    for (int n = 1; n < NCLU; ++n) { float v = simS[t][n]; if (v > best) { best = v; bi = n; } }
    idx[p0 + t] = bi;
  }
}

// ---------------------------------------------------------------------------
// 3) zero accumulators
// ---------------------------------------------------------------------------
__global__ void k_zero(float* __restrict__ sumx, float* __restrict__ cnt) {
  int i = blockIdx.x * 256 + threadIdx.x;
  if (i < NB * NCLU * CC) sumx[i] = 0.f;
  if (i < NB * NCLU)      cnt[i]  = 0.f;
}

// ---------------------------------------------------------------------------
// 4) scatter-add per pixel into (b, cluster, c) and counts
// ---------------------------------------------------------------------------
__global__ __launch_bounds__(256) void k_scatter(const float* __restrict__ x,
                                                 const int* __restrict__ idx,
                                                 float* __restrict__ sumx,
                                                 float* __restrict__ cnt) {
  int p = blockIdx.x, t = threadIdx.x;
  int b = p / HW, s = p % HW;
  int n = idx[p];
  float* dst = sumx + ((size_t)(b * NCLU + n)) * CC;
  const float* src = x + (size_t)b * CC * HW + s;
  for (int c = t; c < CC; c += 256) atomicAdd(&dst[c], src[(size_t)c * HW]);
  if (t == 0) atomicAdd(&cnt[b * NCLU + n], 1.0f);
}

// ---------------------------------------------------------------------------
// 5) centroids_local = sum_x / max(count, 1)   (in place)
// ---------------------------------------------------------------------------
__global__ void k_final(float* __restrict__ sumx, const float* __restrict__ cnt) {
  int i = blockIdx.x * 256 + threadIdx.x;
  if (i < NB * NCLU * CC) sumx[i] = sumx[i] / fmaxf(cnt[i >> 10], 1.0f);
}

// ---------------------------------------------------------------------------
// 6) alpha + y (bf16, row-major [pixel][C] as uint pairs) -- 128 pixels/block
// ---------------------------------------------------------------------------
__global__ __launch_bounds__(256) void k_y(const float* __restrict__ x,
                                           const int* __restrict__ idx,
                                           const float* __restrict__ cl,
                                           unsigned int* __restrict__ ybf) {
  __shared__ float red[256];
  __shared__ float alphas[128];
  __shared__ int   ns[128];
  int t = threadIdx.x;
  int p0 = blockIdx.x * 128;
  int b = p0 / HW, s0 = p0 % HW;
  if (t < 128) ns[t] = idx[p0 + t];
  __syncthreads();

  // phase 1: sum of delta^2 per pixel (coalesced over pixels)
  int pix = t & 127, half = t >> 7;
  const float* xb  = x + (size_t)b * CC * HW + s0 + pix;
  const float* clr = cl + ((size_t)b * NCLU + ns[pix]) * CC;
  float partial = 0.f;
  for (int c = half; c < CC; c += 2) {
    float d = clr[c] - xb[(size_t)c * HW];
    partial += d * d;
  }
  red[t] = partial; __syncthreads();
  if (t < 128) alphas[t] = expf(-(red[t] + red[t + 128]) * (1.0f / 1024.0f));
  __syncthreads();

  // phase 2: y = x + alpha*delta -> bf16 pairs, row-major per pixel
  int pix2 = t >> 1;
  float a = alphas[pix2];
  const float* xb2  = x + (size_t)b * CC * HW + s0 + pix2;
  const float* clr2 = cl + ((size_t)b * NCLU + ns[pix2]) * CC;
  unsigned int* yrow = ybf + (size_t)(p0 + pix2) * (CC / 2);
  for (int j = 0; j < 256; ++j) {
    int cp = (t & 1) * 256 + j;
    int c = cp * 2;
    float x0 = xb2[(size_t)c * HW], x1 = xb2[(size_t)(c + 1) * HW];
    float y0 = x0 + a * (clr2[c] - x0);
    float y1 = x1 + a * (clr2[c + 1] - x1);
    yrow[cp] = (unsigned)f2bf(y0) | ((unsigned)f2bf(y1) << 16);
  }
}

// ---------------------------------------------------------------------------
// 7) pack W -> bf16 [c/2][o] uint pairs (WMMA B layout)
// ---------------------------------------------------------------------------
__global__ void k_packW(const float* __restrict__ W, unsigned int* __restrict__ Wpk) {
  int i = blockIdx.x * 256 + threadIdx.x;          // pair id, 1024*512 total
  int o  = i >> 9;
  int cp = i & 511;
  int c = cp * 2;
  unsigned lo = f2bf(W[(size_t)o * CC + c]);
  unsigned hi = f2bf(W[(size_t)o * CC + c + 1]);
  Wpk[(size_t)cp * CC + o] = lo | (hi << 16);
}

// ---------------------------------------------------------------------------
// 8) out = relu(W @ y + b) -- bf16 WMMA GEMM, async global->LDS double buffer
// ---------------------------------------------------------------------------
__global__ __launch_bounds__(256) void k_gemm(const unsigned int* __restrict__ yb,
                                              const unsigned int* __restrict__ Wpk,
                                              const float* __restrict__ bias,
                                              float* __restrict__ out) {
  // row strides chosen so every async B128 destination is 16B-aligned and the
  // fragment-read bank patterns stay conflict-free (gcd(20,64)=4 -> distinct).
  __shared__ unsigned int As[2][128][20];   // [buf][pixel][K-pair], 80B rows
  __shared__ unsigned int Bs[2][16][132];   // [buf][K-pair][col],  528B rows
  __shared__ float outT[128][33];           // epilogue transpose staging

  int t = threadIdx.x, lane = t & 31, wid = t >> 5;
  int p0 = blockIdx.x * 128;
  int o0 = blockIdx.y * 128;
  int wm = wid & 3, wn = wid >> 2;          // 4x2 wave grid; wave tile 32(M)x64(N)

  v8f acc[2][4];
#pragma unroll
  for (int it = 0; it < 2; ++it)
#pragma unroll
    for (int jt = 0; jt < 4; ++jt) acc[it][jt] = vzero8();

  int am = t >> 1;            // A stage: row
  int aj = (t & 1) * 8;       //          pair chunk (0 or 8)
  int br = t >> 4;            // B stage: K-pair row
  int bn = (t & 15) * 8;      //          col chunk

  // per-thread async destinations/sources (byte offsets)
  unsigned aLds0 = ldsoff(&As[0][am][aj]);
  unsigned aLds1 = ldsoff(&As[1][am][aj]);
  unsigned bLds0 = ldsoff(&Bs[0][br][bn]);
  unsigned bLds1 = ldsoff(&Bs[1][br][bn]);
  unsigned aG = (unsigned)(((size_t)(p0 + am) * 512 + aj) * 4);  // + kt*64 later
  unsigned bG = (unsigned)(((size_t)br * CC + o0 + bn) * 4);     // + kt*16*4096 later

  const int NKT = CC / 32;    // 32 K-tiles

  // prologue: stage tile 0 into buffer 0
  {
    async_cp16(aLds0, yb, aG);
    async_cp16(aLds0 + 16, yb, aG + 16);
    async_cp16(bLds0, Wpk, bG);
    async_cp16(bLds0 + 16, Wpk, bG + 16);
  }

  for (int kt = 0; kt < NKT; ++kt) {
    wait_async0();            // tile kt landed in LDS (only tile in flight)
    __syncthreads();          // visible to all waves; prev compute finished

    if (kt + 1 < NKT) {       // overlap: stage kt+1 while computing kt
      int nb = (kt + 1) & 1;
      unsigned aL = nb ? aLds1 : aLds0;
      unsigned bL = nb ? bLds1 : bLds0;
      unsigned aOff = aG + (unsigned)(kt + 1) * 64;        // 16 pairs * 4B
      unsigned bOff = bG + (unsigned)(kt + 1) * 16 * 4096; // 16 rows * 1024 uints
      async_cp16(aL, yb, aOff);
      async_cp16(aL + 16, yb, aOff + 16);
      async_cp16(bL, Wpk, bOff);
      async_cp16(bL + 16, Wpk, bOff + 16);
    }

    int cb = kt & 1;
    int khp = (lane >> 4) * 4;
    int kbp = (lane >> 4) * 4;
    v16bf afr[2];
#pragma unroll
    for (int it = 0; it < 2; ++it) {
      v8u au;
      int m = wm * 32 + it * 16 + (lane & 15);
#pragma unroll
      for (int r = 0; r < 4; ++r) au[r]     = As[cb][m][khp + r];
#pragma unroll
      for (int r = 0; r < 4; ++r) au[4 + r] = As[cb][m][8 + khp + r];
      afr[it] = __builtin_bit_cast(v16bf, au);
    }
#pragma unroll
    for (int jt = 0; jt < 4; ++jt) {
      v8u bu;
      int n = wn * 64 + jt * 16 + (lane & 15);
#pragma unroll
      for (int r = 0; r < 8; ++r) bu[r] = Bs[cb][kbp + (r & 3) + ((r >> 2) << 3)][n];
      v16bf bfr = __builtin_bit_cast(v16bf, bu);
#pragma unroll
      for (int it = 0; it < 2; ++it)
        acc[it][jt] = __builtin_amdgcn_wmma_f32_16x16x32_bf16(
            false, afr[it], false, bfr, (short)0, acc[it][jt], false, false);
    }
  }

  // epilogue: LDS transpose in 4 chunks of 32 output columns, bias+relu,
  // coalesced channel-major store
  int bI = p0 / HW, s0 = p0 % HW;
  for (int ch = 0; ch < 4; ++ch) {
    __syncthreads();
    if (wn == (ch >> 1)) {
      int jt0 = (ch & 1) * 2;
#pragma unroll
      for (int jj = 0; jj < 2; ++jj) {
        int jt = jt0 + jj;
        int ncol = jj * 16 + (lane & 15);
#pragma unroll
        for (int it = 0; it < 2; ++it) {
          int mb = wm * 32 + it * 16 + ((lane >> 4) << 3);
#pragma unroll
          for (int v = 0; v < 8; ++v) outT[mb + v][ncol] = acc[it][jt][v];
        }
      }
    }
    __syncthreads();
    int n  = t >> 3;             // 0..31 local col
    int pg = (t & 7) * 16;       // pixel start
    int o  = o0 + ch * 32 + n;
    float bv = bias[o];
    float* orow = out + ((size_t)bI * CC + o) * HW + s0;
    for (int i = 0; i < 16; ++i) {
      float v = outT[pg + i][n] + bv;
      orow[pg + i] = v > 0.f ? v : 0.f;
    }
  }
}

// ---------------------------------------------------------------------------
extern "C" void kernel_launch(void* const* d_in, const int* in_sizes, int n_in,
                              void* d_out, int out_size, void* d_ws, size_t ws_size,
                              hipStream_t stream) {
  (void)in_sizes; (void)n_in; (void)out_size; (void)ws_size;
  const float* x    = (const float*)d_in[0];
  const float* cent = (const float*)d_in[1];
  const float* W    = (const float*)d_in[2];
  const float* bias = (const float*)d_in[3];
  float* out = (float*)d_out;

  char* ws = (char*)d_ws;
  constexpr size_t OFF_CNP = 0;
  constexpr size_t OFF_IDX = OFF_CNP + (size_t)512 * NCLU * 4;       // 163840
  constexpr size_t OFF_SUM = OFF_IDX + (size_t)NPIX * 4;             // +196608
  constexpr size_t OFF_CNT = OFF_SUM + (size_t)NB * NCLU * CC * 4;   // +2621440
  constexpr size_t OFF_WPK = OFF_CNT + 4096;
  constexpr size_t OFF_YBF = OFF_WPK + (size_t)512 * CC * 4;         // +2097152

  unsigned int* cnP = (unsigned int*)(ws + OFF_CNP);
  int*          idx = (int*)(ws + OFF_IDX);
  float*        sumx = (float*)(ws + OFF_SUM);
  float*        cnt  = (float*)(ws + OFF_CNT);
  unsigned int* Wpk  = (unsigned int*)(ws + OFF_WPK);
  unsigned int* ybf  = (unsigned int*)(ws + OFF_YBF);

  k_cent   <<<NCLU, 256, 0, stream>>>(cent, cnP);
  k_sim    <<<NPIX / 128, 256, 0, stream>>>(x, cnP, idx);
  k_zero   <<<(NB * NCLU * CC + 255) / 256, 256, 0, stream>>>(sumx, cnt);
  k_scatter<<<NPIX, 256, 0, stream>>>(x, idx, sumx, cnt);
  k_final  <<<(NB * NCLU * CC + 255) / 256, 256, 0, stream>>>(sumx, cnt);
  k_y      <<<NPIX / 128, 256, 0, stream>>>(x, idx, sumx, ybf);
  k_packW  <<<(CC * 512 + 255) / 256, 256, 0, stream>>>(W, Wpk);
  dim3 gg(NPIX / 128, CC / 128);
  k_gemm   <<<gg, 256, 0, stream>>>(ybf, Wpk, bias, out);
}
